// MultiScaleTemporalGraphConv_33827162423950
// MI455X (gfx1250) — matrix operand order repaired and checked by reading, not compiled, not run on hardware
//
#include <hip/hip_runtime.h>
#include <math.h>

typedef __attribute__((ext_vector_type(16))) _Float16 v16h;
typedef __attribute__((ext_vector_type(8)))  float    v8f;

#define CIN_  64
#define T_    32
#define HC_   128
#define NHEAD 4
#define PC_   32
#define NEGS  0.2f
#define NEG_INF (-3.402823e38f)

// ---------------- WMMA tiled GEMM: C[M,N] = A[M,K](f16) @ B(f16,pre-swizzled) ------------
// Each wave computes a 16x32 slab (two 16x16 WMMA tiles sharing the A fragment).
// B is stored in exact WMMA fragment order: [kchunk][colTile16][lane][16 halves],
// so each B fragment is one contiguous 32-byte v16h load.
template<int OUT_MODE> // 0=f32, 1=f16, 2=f32+relu
__global__ __launch_bounds__(128)
void gemm_wmma_kernel(const _Float16* __restrict__ A, int lda,
                      const _Float16* __restrict__ Bsw, int colTiles16, int ncols,
                      const float* __restrict__ bias,
                      void* __restrict__ C, int ldc, int K)
{
  const int lane = threadIdx.x & 31;
  const int wave = threadIdx.x >> 5;
  const int col0 = (blockIdx.y * 4 + wave) * 32;      // 32 columns per wave
  if (col0 >= ncols) return;
  const int ct0  = col0 >> 4;                         // first 16-col tile index
  const int row0 = blockIdx.x * 16;
  const int m    = lane & 15;
  const int half = lane >> 4;

  v8f c0 = {0.f,0.f,0.f,0.f,0.f,0.f,0.f,0.f};
  v8f c1 = {0.f,0.f,0.f,0.f,0.f,0.f,0.f,0.f};
  const _Float16* Arow = A + (size_t)(row0 + m) * lda;
#pragma unroll 2
  for (int kk = 0; kk < K; kk += 32) {
    const _Float16* ap = Arow + kk;
    __builtin_prefetch(ap + 32, 0, 1);                // next K-chunk -> global_prefetch_b8
    v16h af;
#pragma unroll
    for (int e = 0; e < 16; ++e) {                    // A 16-bit 16x32 fragment (contiguous k = half*16 + e)
      af[e] = ap[half * 16 + e];
    }
    const size_t bbase = ((size_t)(kk >> 5) * colTiles16 + ct0) * 512 + (size_t)lane * 16;
    v16h bf0 = *(const v16h*)(Bsw + bbase);           // fragment for col tile ct0
    v16h bf1 = *(const v16h*)(Bsw + bbase + 512);     // fragment for col tile ct0+1
    c0 = __builtin_amdgcn_wmma_f32_16x16x32_f16(false, af, false, bf0, (short)0, c0, false, false);
    c1 = __builtin_amdgcn_wmma_f32_16x16x32_f16(false, af, false, bf1, (short)0, c1, false, false);
  }
#pragma unroll
  for (int d = 0; d < 8; ++d) {                       // C/D layout: M = d + 8*half, N = lane%16
    const int r = row0 + d + half * 8;
#pragma unroll
    for (int q = 0; q < 2; ++q) {
      const int cc = col0 + q * 16 + m;
      float v = (q ? c1[d] : c0[d]) + (bias ? bias[cc] : 0.f);
      if (OUT_MODE == 2) v = v > 0.f ? v : 0.f;
      if (OUT_MODE == 1) ((_Float16*)C)[(size_t)r * ldc + cc] = (_Float16)v;
      else               ((float*)C)[(size_t)r * ldc + cc] = v;
    }
  }
}

// ---------------- Weight prep: folded conv weights + fragment-swizzled B matrices ---------
// Swizzled layout: element t -> e = t%16, lane = (t/16)%32, ct = (t/512)%colTiles16,
// kchunk = t/(512*colTiles16); logical k = kchunk*32 + (lane>>4)*16 + e, n = ct*16 + (lane&15).
__global__ void prep_kernel(const float* __restrict__ conv_w, const float* __restrict__ lin_w,
                            const float* __restrict__ fuse_w,
                            _Float16* __restrict__ WeffT, _Float16* __restrict__ linT,
                            _Float16* __restrict__ fuseT)
{
  int t = blockIdx.x * blockDim.x + threadIdx.x;
  const float invT = 1.0f / (float)T_;
  if (t < 448 * 96) {                                  // Weff: K=448, ncols=96 (colTiles16=6)
    int e = t & 15, lane = (t >> 4) & 31, ct = (t >> 9) % 6, kchunk = t / (512 * 6);
    int k = kchunk * 32 + (lane >> 4) * 16 + e;
    int n = ct * 16 + (lane & 15);
    int c = k / 7, j = k % 7;
    int s = n / 32, pc = n % 32;
    const float* w = conv_w + ((size_t)(s * 32 + pc) * CIN_ + c) * 3;
    float v = 0.f;
    if (j == 0)          v = (w[0] + w[1] + w[2]) * invT;   // weight on S
    else if (j == 1 + s) v = -w[0] * invT;                  // weight on tail_r
    else if (j == 4 + s) v = -w[2] * invT;                  // weight on head_r
    WeffT[t] = (_Float16)v;
    return;
  }
  t -= 448 * 96;
  if (t < 3 * 32 * 128) {                              // lin (per scale): K=32, ncols=128 (8 tiles)
    int e = t & 15, lane = (t >> 4) & 31, ct = (t >> 9) % 8, s = t / 4096;
    int k = (lane >> 4) * 16 + e;
    int n = ct * 16 + (lane & 15);
    linT[t] = (_Float16)lin_w[((size_t)s * 128 + n) * 32 + k];
    return;
  }
  t -= 3 * 32 * 128;
  if (t < 384 * 96) {                                  // fuse: K=384, ncols=96 (6 tiles)
    int e = t & 15, lane = (t >> 4) & 31, ct = (t >> 9) % 6, kchunk = t / (512 * 6);
    int k = kchunk * 32 + (lane >> 4) * 16 + e;
    int o = ct * 16 + (lane & 15);
    fuseT[t] = (_Float16)fuse_w[(size_t)o * 384 + k];
  }
}

// ---------------- Temporal reductions: per (n,cin) -> S, tail1/2/4, head1/2/4 -------------
__global__ void feat_kernel(const float* __restrict__ x, _Float16* __restrict__ F, int NC)
{
  int t = blockIdx.x * blockDim.x + threadIdx.x;
  if (t >= NC) return;
  const float* p = x + (size_t)t * T_;
  float S = 0.f;
#pragma unroll
  for (int i = 0; i < T_; ++i) S += p[i];
  float h1 = p[0], h2 = h1 + p[1], h4 = h2 + p[2] + p[3];
  float t1 = p[31], t2 = t1 + p[30], t4 = t2 + p[29] + p[28];
  _Float16* f = F + (size_t)t * 7;
  f[0] = (_Float16)S;
  f[1] = (_Float16)t1; f[2] = (_Float16)t2; f[3] = (_Float16)t4;
  f[4] = (_Float16)h1; f[5] = (_Float16)h2; f[6] = (_Float16)h4;
}

// ---------------- CSR build ---------------------------------------------------------------
__global__ void deg_kernel(const long long* __restrict__ e0, int E, int Etot, int* __restrict__ deg)
{
  int e = blockIdx.x * blockDim.x + threadIdx.x;
  if (e >= Etot) return;
  int row = (e < E) ? (int)e0[e] : (e - E);
  atomicAdd(&deg[row], 1);
}

__global__ void scan_kernel(const int* __restrict__ deg, int* __restrict__ offs, int n)
{
  __shared__ int sh[1024];
  __shared__ int carry;
  if (threadIdx.x == 0) carry = 0;
  __syncthreads();
  for (int base = 0; base < n; base += 1024) {
    int i = base + threadIdx.x;
    int v = (i < n) ? deg[i] : 0;
    sh[threadIdx.x] = v; __syncthreads();
    for (int o = 1; o < 1024; o <<= 1) {
      int tv = (threadIdx.x >= o) ? sh[threadIdx.x - o] : 0;
      __syncthreads();
      sh[threadIdx.x] += tv;
      __syncthreads();
    }
    if (i < n) offs[i] = carry + sh[threadIdx.x] - v;  // exclusive
    __syncthreads();
    if (threadIdx.x == 1023) carry += sh[1023];
    __syncthreads();
  }
}

__global__ void fill_kernel(const long long* __restrict__ e0, int E, int Etot,
                            const int* __restrict__ offs, int* __restrict__ cursor,
                            int* __restrict__ elist)
{
  int e = blockIdx.x * blockDim.x + threadIdx.x;
  if (e >= Etot) return;
  int row = (e < E) ? (int)e0[e] : (e - E);
  int pos = atomicAdd(&cursor[row], 1);
  elist[offs[row] + pos] = e;
}

// ---------------- Attention dot products --------------------------------------------------
__global__ void att_kernel(const float* __restrict__ Hm, const float* __restrict__ a_s,
                           const float* __restrict__ a_d, float* __restrict__ asrc,
                           float* __restrict__ adst, int N)
{
  int t = blockIdx.x * blockDim.x + threadIdx.x;
  if (t >= N * NHEAD) return;
  int n = t >> 2, h = t & 3;
  const float* hv = Hm + (size_t)n * HC_ + h * PC_;
  float s1 = 0.f, s2 = 0.f;
#pragma unroll
  for (int pc = 0; pc < PC_; ++pc) { s1 += hv[pc] * a_s[h * PC_ + pc]; s2 += hv[pc] * a_d[h * PC_ + pc]; }
  asrc[t] = s1; adst[t] = s2;
}

// ---------------- Per-row softmax entropy (CSR, no atomics) -------------------------------
__global__ void entropy_kernel(const int* __restrict__ offs, const int* __restrict__ deg,
                               const int* __restrict__ elist, const long long* __restrict__ e1,
                               int E, int N, const float* __restrict__ asrc,
                               const float* __restrict__ adst, float* __restrict__ ent)
{
  int n = blockIdx.x * blockDim.x + threadIdx.x;
  if (n >= N) return;
  int o = offs[n], d = deg[n];
  float m[NHEAD] = {NEG_INF, NEG_INF, NEG_INF, NEG_INF};
  float as[NHEAD];
#pragma unroll
  for (int h = 0; h < NHEAD; ++h) as[h] = asrc[n * NHEAD + h];
  for (int i = 0; i < d; ++i) {
    int e = elist[o + i]; int col = (e < E) ? (int)e1[e] : (e - E);
#pragma unroll
    for (int h = 0; h < NHEAD; ++h) { float a = as[h] + adst[col * NHEAD + h]; m[h] = fmaxf(m[h], a); }
  }
  float ss[NHEAD] = {0.f, 0.f, 0.f, 0.f};
  for (int i = 0; i < d; ++i) {
    int e = elist[o + i]; int col = (e < E) ? (int)e1[e] : (e - E);
#pragma unroll
    for (int h = 0; h < NHEAD; ++h) ss[h] += expf(as[h] + adst[col * NHEAD + h] - m[h]);
  }
  float ev = 0.f;
  for (int i = 0; i < d; ++i) {
    int e = elist[o + i]; int col = (e < E) ? (int)e1[e] : (e - E);
#pragma unroll
    for (int h = 0; h < NHEAD; ++h) {
      float p = expf(as[h] + adst[col * NHEAD + h] - m[h]) / (ss[h] + 1e-16f);
      ev += -(p * logf(p + 1e-8f));
    }
  }
  ent[n] = ev;
}

// ---------------- Adaptive sparsity scalar ------------------------------------------------
__global__ void sp_kernel(const int* __restrict__ deg, const float* __restrict__ ent, int N,
                          float* __restrict__ sp_out)
{
  __shared__ float sc[1024], se[1024];
  float c = 0.f, s = 0.f;
  for (int i = threadIdx.x; i < N; i += 1024)
    if (deg[i] > 1) { c += 1.f; s += ent[i]; }
  sc[threadIdx.x] = c; se[threadIdx.x] = s; __syncthreads();
  for (int o = 512; o > 0; o >>= 1) {
    if (threadIdx.x < o) { sc[threadIdx.x] += sc[threadIdx.x + o]; se[threadIdx.x] += se[threadIdx.x + o]; }
    __syncthreads();
  }
  if (threadIdx.x == 0) {
    float cnt = sc[0];
    float mean_ent = se[0] / fmaxf(cnt, 1.0f);
    float norm_ent = mean_ent / (logf(fmaxf(cnt, 2.0f)) + 1e-8f);
    sp_out[0] = fminf(1.0f, 0.1f * (1.0f + norm_ent));
  }
}

// ---------------- Top-k sparsify + leaky-relu + second softmax (exact lexsort ranks) ------
__global__ void alpha_kernel(const int* __restrict__ offs, const int* __restrict__ deg,
                             const int* __restrict__ elist, const long long* __restrict__ e1,
                             int E, int N, const float* __restrict__ asrc,
                             const float* __restrict__ adst, const float* __restrict__ sp_ptr,
                             float* __restrict__ alpha)
{
  int n = blockIdx.x * blockDim.x + threadIdx.x;
  if (n >= N) return;
  const float sp = sp_ptr[0];
  int o = offs[n], d = deg[n];
  int kthr = max(1, (int)floorf((float)d * sp));
  for (int h = 0; h < NHEAD; ++h) {
    float as = asrc[n * NHEAD + h];
    float m = NEG_INF;
    for (int i = 0; i < d; ++i) {
      int e = elist[o + i]; int col = (e < E) ? (int)e1[e] : (e - E);
      float ai = as + adst[col * NHEAD + h];
      int rank = 0;
      for (int jx = 0; jx < d; ++jx) {
        if (jx == i) continue;
        int e2 = elist[o + jx]; int c2 = (e2 < E) ? (int)e1[e2] : (e2 - E);
        float aj = as + adst[c2 * NHEAD + h];
        if (aj > ai || (aj == ai && e2 < e)) rank++;   // stable-sort tie-break by edge id
      }
      float v = (rank < kthr) ? ai : 0.0f;
      float lr = v > 0.f ? v : NEGS * v;
      alpha[(size_t)e * NHEAD + h] = lr;
      m = fmaxf(m, lr);
    }
    float ss = 0.f;
    for (int i = 0; i < d; ++i) {
      int e = elist[o + i];
      ss += expf(alpha[(size_t)e * NHEAD + h] - m);
    }
    for (int i = 0; i < d; ++i) {
      int e = elist[o + i];
      alpha[(size_t)e * NHEAD + h] = expf(alpha[(size_t)e * NHEAD + h] - m) / (ss + 1e-16f);
    }
  }
}

// ---------------- Edge message scatter (hardware FP atomics) ------------------------------
__global__ void msg_kernel(const long long* __restrict__ e0, const long long* __restrict__ e1,
                           int E, long long total, const float* __restrict__ alpha,
                           const float* __restrict__ Hm, float* __restrict__ acc)
{
  long long t = (long long)blockIdx.x * blockDim.x + threadIdx.x;
  if (t >= total) return;
  int e = (int)(t >> 7);
  int j = (int)(t & 127);
  int row = (e < E) ? (int)e0[e] : (e - E);
  int col = (e < E) ? (int)e1[e] : (e - E);
  float a = alpha[(size_t)e * NHEAD + (j >> 5)];
  unsafeAtomicAdd(&acc[(size_t)col * HC_ + j], a * Hm[(size_t)row * HC_ + j]);
}

// ---------------- Bias + LayerNorm (one wave per node), write f16 concat ------------------
__global__ void ln_kernel(const float* __restrict__ acc, const float* __restrict__ gbias,
                          const float* __restrict__ lw, const float* __restrict__ lb,
                          _Float16* __restrict__ cat, int N, int s)
{
  int wid = (blockIdx.x * blockDim.x + threadIdx.x) >> 5;
  int lane = threadIdx.x & 31;
  if (wid >= N) return;
  float v[4]; float sum = 0.f;
#pragma unroll
  for (int q = 0; q < 4; ++q) {
    int j = lane * 4 + q;
    v[q] = acc[(size_t)wid * HC_ + j] + gbias[j];
    sum += v[q];
  }
#pragma unroll
  for (int o = 16; o > 0; o >>= 1) sum += __shfl_xor(sum, o, 32);
  float mu = sum * (1.0f / HC_);
  float var = 0.f;
#pragma unroll
  for (int q = 0; q < 4; ++q) { float dd = v[q] - mu; var += dd * dd; }
#pragma unroll
  for (int o = 16; o > 0; o >>= 1) var += __shfl_xor(var, o, 32);
  var *= (1.0f / HC_);
  float inv = rsqrtf(var + 1e-5f);
#pragma unroll
  for (int q = 0; q < 4; ++q) {
    int j = lane * 4 + q;
    cat[(size_t)wid * 384 + s * HC_ + j] = (_Float16)(lw[j] * (v[q] - mu) * inv + lb[j]);
  }
}

// =========================================================================================
extern "C" void kernel_launch(void* const* d_in, const int* in_sizes, int n_in,
                              void* d_out, int out_size, void* d_ws, size_t ws_size,
                              hipStream_t stream)
{
  (void)n_in; (void)out_size; (void)ws_size;
  const float*     x        = (const float*)d_in[0];
  const long long* eidx     = (const long long*)d_in[1];   // int64 per reference
  const float*     conv_w   = (const float*)d_in[2];
  const float*     conv_b   = (const float*)d_in[3];
  const float*     lin_w    = (const float*)d_in[4];
  const float*     att_src  = (const float*)d_in[5];
  const float*     att_dst  = (const float*)d_in[6];
  const float*     gat_bias = (const float*)d_in[7];
  const float*     ln_w     = (const float*)d_in[8];
  const float*     ln_b     = (const float*)d_in[9];
  const float*     fuse_w   = (const float*)d_in[10];
  const float*     fuse_b   = (const float*)d_in[11];

  const int N    = in_sizes[0] / (CIN_ * T_);
  const int E    = in_sizes[1] / 2;
  const int Etot = E + N;
  const long long* e0 = eidx;
  const long long* e1 = eidx + E;

  // ---- workspace carve ----
  char* base = (char*)d_ws; size_t off = 0;
  auto take = [&](size_t b) -> void* { void* r = base + off; off += (b + 255) & ~(size_t)255; return r; };
  _Float16* F      = (_Float16*)take((size_t)N * 448 * sizeof(_Float16));
  _Float16* WeffT  = (_Float16*)take((size_t)448 * 96 * sizeof(_Float16));
  _Float16* linT   = (_Float16*)take((size_t)3 * 32 * 128 * sizeof(_Float16));
  _Float16* fuseT  = (_Float16*)take((size_t)384 * 96 * sizeof(_Float16));
  _Float16* hp     = (_Float16*)take((size_t)N * 96 * sizeof(_Float16));
  float*    Hm     = (float*)take((size_t)N * HC_ * sizeof(float));
  float*    asrc   = (float*)take((size_t)N * NHEAD * sizeof(float));
  float*    adst   = (float*)take((size_t)N * NHEAD * sizeof(float));
  int*      deg    = (int*)take((size_t)N * sizeof(int));
  int*      offs   = (int*)take((size_t)N * sizeof(int));
  int*      cursor = (int*)take((size_t)N * sizeof(int));
  int*      elist  = (int*)take((size_t)Etot * sizeof(int));
  float*    ent    = (float*)take((size_t)N * sizeof(float));
  float*    spv    = (float*)take(256);
  float*    alpha  = (float*)take((size_t)Etot * NHEAD * sizeof(float));
  float*    acc    = (float*)take((size_t)N * HC_ * sizeof(float));
  _Float16* cat    = (_Float16*)take((size_t)N * 384 * sizeof(_Float16));

  const int rowTiles = N / 16;   // N = 50000 = 3125 * 16

  // ---- graph CSR (scale-independent) ----
  hipMemsetAsync(deg, 0, (size_t)N * sizeof(int), stream);
  hipMemsetAsync(cursor, 0, (size_t)N * sizeof(int), stream);
  deg_kernel<<<(Etot + 255) / 256, 256, 0, stream>>>(e0, E, Etot, deg);
  scan_kernel<<<1, 1024, 0, stream>>>(deg, offs, N);
  fill_kernel<<<(Etot + 255) / 256, 256, 0, stream>>>(e0, E, Etot, offs, cursor, elist);

  // ---- weights + temporal features + folded conv GEMM ----
  prep_kernel<<<(448 * 96 + 3 * 32 * 128 + 384 * 96 + 255) / 256, 256, 0, stream>>>(
      conv_w, lin_w, fuse_w, WeffT, linT, fuseT);
  feat_kernel<<<(N * CIN_ + 255) / 256, 256, 0, stream>>>(x, F, N * CIN_);
  // hp(N x 96, f16) = F(N x 448) @ Weff(448 x 96) + conv_b   (96 cols = 3 waves of 32)
  gemm_wmma_kernel<1><<<dim3(rowTiles, 1), 128, 0, stream>>>(
      F, 448, WeffT, 6, 96, conv_b, hp, 96, 448);

  // ---- per-scale GAT pipeline ----
  for (int s = 0; s < 3; ++s) {
    // H(N x 128, f32) = hp[:, s*32:(s+1)*32] @ lin_s(32 x 128)   (128 cols = 4 waves of 32)
    gemm_wmma_kernel<0><<<dim3(rowTiles, 1), 128, 0, stream>>>(
        hp + s * 32, 96, linT + (size_t)s * 32 * 128, 8, 128, nullptr, Hm, 128, 32);
    att_kernel<<<(N * NHEAD + 255) / 256, 256, 0, stream>>>(
        Hm, att_src + s * NHEAD * PC_, att_dst + s * NHEAD * PC_, asrc, adst, N);
    entropy_kernel<<<(N + 255) / 256, 256, 0, stream>>>(offs, deg, elist, e1, E, N, asrc, adst, ent);
    sp_kernel<<<1, 1024, 0, stream>>>(deg, ent, N, spv);
    alpha_kernel<<<(N + 255) / 256, 256, 0, stream>>>(offs, deg, elist, e1, E, N, asrc, adst, spv, alpha);
    hipMemsetAsync(acc, 0, (size_t)N * HC_ * sizeof(float), stream);
    {
      long long total = (long long)Etot * HC_;
      int blocks = (int)((total + 255) / 256);
      msg_kernel<<<blocks, 256, 0, stream>>>(e0, e1, E, total, alpha, Hm, acc);
    }
    ln_kernel<<<(N * 32 + 255) / 256, 256, 0, stream>>>(
        acc, gat_bias + s * HC_, ln_w + s * HC_, ln_b + s * HC_, cat, N, s);
  }

  // ---- fusion: out(N x 96, f32) = relu(cat(N x 384) @ fuse(384 x 96) + fuse_b) ----
  gemm_wmma_kernel<2><<<dim3(rowTiles, 1), 128, 0, stream>>>(
      cat, 384, fuseT, 6, 96, fuse_b, d_out, 96, 384);
}
